// AllModel_35828617183965
// MI455X (gfx1250) — compile-verified, matrix operand order checked by reading, not run on hardware
//
#include <hip/hip_runtime.h>

typedef __attribute__((ext_vector_type(16))) __bf16 v16bf;
typedef __attribute__((ext_vector_type(8)))  __bf16 bf16x8;
typedef __attribute__((ext_vector_type(8)))  float  v8f;
typedef __attribute__((ext_vector_type(4)))  float  v4f;

#define B_EP   32
#define NSHOT  200
#define NQRY   100
#define CIN    8
#define LIN    512
#define H1     32
#define H2     64
#define H3     128
#define FDIM   128
#define NWAY   20
#define NSAMP  (B_EP * (NSHOT + NQRY))   /* 9600 */
#define NSUP   (B_EP * NSHOT)            /* 6400 */
#define EPS_BN 1e-5f

__device__ __forceinline__ v16bf cat8(bf16x8 a, bf16x8 b) {
    return __builtin_shufflevector(a, b, 0,1,2,3,4,5,6,7,8,9,10,11,12,13,14,15);
}

__device__ __forceinline__ v8f wmma_bf16(v16bf a, v16bf b, v8f c) {
    return __builtin_amdgcn_wmma_f32_16x16x32_bf16(false, a, false, b,
                                                   (short)0, c, false, false);
}

// swap adjacent lanes (0<->1, 2<->3, ...) via DPP quad_perm:[1,0,3,2]
__device__ __forceinline__ float dpp_swap1(float x) {
    return __int_as_float(__builtin_amdgcn_update_dpp(
        0, __float_as_int(x), 0xB1, 0xF, 0xF, true));
}

// ---------------------------------------------------------------------------
// One conv stage as implicit GEMM with K ordered (tap k, channel c).
//  inT : position-major activations [(LI+2) x CI], rows 0 and LI+1 are zero halo
//  wp  : weights [CO x 3*CI], row o holds K = k*CI + c
//  out : pooled activations, position-major (bf16 halo layout or f32 flat)
// ---------------------------------------------------------------------------
template<int CI, int CO, int LI, bool FOUT>
__device__ __forceinline__ void conv_stage(
    const __bf16* __restrict__ inT,
    const __bf16* __restrict__ wp,
    const float*  __restrict__ sc, const float* __restrict__ sh,
    __bf16* __restrict__ outT,        // used when !FOUT: [(LI/2+2) x CO]
    float*  __restrict__ outF,        // used when  FOUT: [LI/2 x CO]
    int lane, int wave)
{
    constexpr int KSTEPS = (3 * CI) / 32;
    constexpr int MT = CO / 16;
    constexpr int NT = LI / 16;
    const int nlo = lane & 15;
    const int hi  = lane >> 4;
    for (int job = wave; job < MT * NT; job += 8) {
        const int mt = job % MT, nt = job / MT;
        const int mrow = mt * 16 + nlo;
        const int col  = nt * 16 + nlo;
        v8f acc = {0.f,0.f,0.f,0.f,0.f,0.f,0.f,0.f};
        #pragma unroll
        for (int ks = 0; ks < KSTEPS; ++ks) {
            const int k = (ks * 32) / CI;            // fixed conv tap this step
            const __bf16* wr = wp  + mrow * (3 * CI) + ks * 32;
            const __bf16* br = inT + (col + k) * CI + ((ks * 32) % CI);
            bf16x8 a0 = *(const bf16x8*)(wr + hi * 8);
            bf16x8 a1 = *(const bf16x8*)(wr + 16 + hi * 8);
            bf16x8 b0 = *(const bf16x8*)(br + hi * 8);
            bf16x8 b1 = *(const bf16x8*)(br + 16 + hi * 8);
            acc = wmma_bf16(cat8(a0, a1), cat8(b0, b1), acc);
        }
        const int ob = mt * 16 + 8 * hi;             // 8 consecutive channels
        float pl[8];
        #pragma unroll
        for (int j = 0; j < 8; ++j) {
            float y = fmaxf(acc[j] * sc[ob + j] + sh[ob + j], 0.0f);
            pl[j] = fmaxf(y, dpp_swap1(y));          // maxpool2 across columns
        }
        if ((lane & 1) == 0) {
            if constexpr (FOUT) {
                v4f lo = {pl[0], pl[1], pl[2], pl[3]};
                v4f hf = {pl[4], pl[5], pl[6], pl[7]};
                *(v4f*)(outF + (col >> 1) * CO + ob)     = lo;
                *(v4f*)(outF + (col >> 1) * CO + ob + 4) = hf;
            } else {
                bf16x8 st;
                #pragma unroll
                for (int j = 0; j < 8; ++j) st[j] = (__bf16)pl[j];
                *(bf16x8*)(outT + ((col >> 1) + 1) * CO + ob) = st;
            }
        }
    }
}

// ---------------------------------------------------------------------------
// Fused encoder: conv->BN->ReLU->pool (x3) + global mean, one block per sample.
// ---------------------------------------------------------------------------
__global__ __launch_bounds__(256)
void encoder_conv_kernel(const float* __restrict__ s_img,
                         const float* __restrict__ q_img,
                         const float* __restrict__ w1, const float* __restrict__ b1,
                         const float* __restrict__ g1, const float* __restrict__ be1,
                         const float* __restrict__ m1, const float* __restrict__ v1,
                         const float* __restrict__ w2, const float* __restrict__ b2,
                         const float* __restrict__ g2, const float* __restrict__ be2,
                         const float* __restrict__ m2, const float* __restrict__ v2,
                         const float* __restrict__ w3, const float* __restrict__ b3,
                         const float* __restrict__ g3, const float* __restrict__ be3,
                         const float* __restrict__ m3, const float* __restrict__ v3,
                         float* __restrict__ feat)
{
    __shared__ __align__(16) __bf16 sinT[(LIN + 2) * CIN];   // 8.03 KB pos-major
    __shared__ __align__(16) __bf16 sw1[H1 * 32];            // 2 KB (K pad 24->32)
    __shared__ __align__(16) __bf16 sw2[H2 * 96];            // 12 KB
    __shared__ __align__(16) __bf16 sw3[H3 * 192];           // 48 KB
    __shared__ __align__(16) __bf16 sa1T[(256 + 2) * H1];    // 16.1 KB
    __shared__ __align__(16) __bf16 sa2T[(128 + 2) * H2];    // 16.3 KB
    __shared__ __align__(16) float  sa3[64 * H3];            // 32 KB
    __shared__ __align__(16) __bf16 szero[8];
    __shared__ float sc1[H1], sh1[H1], sc2[H2], sh2[H2], sc3[H3], sh3[H3];

    const int tid  = threadIdx.x;
    const int lane = tid & 31;
    const int wave = tid >> 5;
    const int n    = blockIdx.x;
    const float* src = (n < NSUP) ? (s_img + (size_t)n * CIN * LIN)
                                  : (q_img + (size_t)(n - NSUP) * CIN * LIN);

    // ---- stage inputs / weights / BN constants into LDS ----------------
    for (int i = tid; i < CIN * LIN; i += 256) {             // coalesced read
        int c = i >> 9, l = i & 511;
        sinT[(l + 1) * CIN + c] = (__bf16)src[i];            // transpose to LDS
    }
    for (int i = tid; i < H1 * 32; i += 256) {               // permute + pad
        int o = i >> 5, kk = i & 31, k = kk >> 3, c = kk & 7;
        sw1[i] = (__bf16)((k < 3) ? w1[o * 24 + c * 3 + k] : 0.0f);
    }
    for (int i = tid; i < H2 * 96; i += 256) {               // K = k*32 + c
        int o = i / 96, kk = i - o * 96, k = kk >> 5, c = kk & 31;
        sw2[i] = (__bf16)w2[o * 96 + c * 3 + k];
    }
    for (int i = tid; i < H3 * 192; i += 256) {              // K = k*64 + c
        int o = i / 192, kk = i - o * 192, k = kk >> 6, c = kk & 63;
        sw3[i] = (__bf16)w3[o * 192 + c * 3 + k];
    }
    for (int i = tid; i < H1; i += 256) {
        float sc = g1[i] * rsqrtf(v1[i] + EPS_BN);
        sc1[i] = sc; sh1[i] = (b1[i] - m1[i]) * sc + be1[i];
    }
    for (int i = tid; i < H2; i += 256) {
        float sc = g2[i] * rsqrtf(v2[i] + EPS_BN);
        sc2[i] = sc; sh2[i] = (b2[i] - m2[i]) * sc + be2[i];
    }
    for (int i = tid; i < H3; i += 256) {
        float sc = g3[i] * rsqrtf(v3[i] + EPS_BN);
        sc3[i] = sc; sh3[i] = (b3[i] - m3[i]) * sc + be3[i];
    }
    // zero halos + zero pad block
    for (int i = tid; i < CIN; i += 256) {
        sinT[i] = (__bf16)0.0f; sinT[(LIN + 1) * CIN + i] = (__bf16)0.0f;
    }
    for (int i = tid; i < H1; i += 256) {
        sa1T[i] = (__bf16)0.0f; sa1T[257 * H1 + i] = (__bf16)0.0f;
    }
    for (int i = tid; i < H2; i += 256) {
        sa2T[i] = (__bf16)0.0f; sa2T[129 * H2 + i] = (__bf16)0.0f;
    }
    if (tid < 8) szero[tid] = (__bf16)0.0f;
    __syncthreads();

    // ---- stage 1: [512 x 8] -> pooled [256 x 32]  (K = k*8+c, pad to 32)
    {
        const int nlo = lane & 15, hi = lane >> 4;
        for (int job = wave; job < 2 * 32; job += 8) {
            int mt = job & 1, nt = job >> 1;
            int mrow = mt * 16 + nlo;
            int col  = nt * 16 + nlo;
            const __bf16* wr = sw1 + mrow * 32;
            bf16x8 a0 = *(const bf16x8*)(wr + hi * 8);
            bf16x8 a1 = *(const bf16x8*)(wr + 16 + hi * 8);
            // run1: K=8*hi..8*hi+7 -> tap k=hi, c=0..7, halo row = col+hi
            bf16x8 b0 = *(const bf16x8*)(sinT + (col + hi) * CIN);
            // run2: hi==0 -> K=16..23 (k=2, row col+2); hi==1 -> K pad, zeros
            const __bf16* p1 = hi ? szero : (sinT + (col + 2) * CIN);
            bf16x8 b1 = *(const bf16x8*)p1;
            v8f acc = {0.f,0.f,0.f,0.f,0.f,0.f,0.f,0.f};
            acc = wmma_bf16(cat8(a0, a1), cat8(b0, b1), acc);
            const int ob = mt * 16 + 8 * hi;
            float pl[8];
            #pragma unroll
            for (int j = 0; j < 8; ++j) {
                float y = fmaxf(acc[j] * sc1[ob + j] + sh1[ob + j], 0.0f);
                pl[j] = fmaxf(y, dpp_swap1(y));
            }
            if ((lane & 1) == 0) {
                bf16x8 st;
                #pragma unroll
                for (int j = 0; j < 8; ++j) st[j] = (__bf16)pl[j];
                *(bf16x8*)(sa1T + ((col >> 1) + 1) * H1 + ob) = st;
            }
        }
    }
    __syncthreads();

    // ---- stage 2: [256 x 32] -> pooled [128 x 64]
    conv_stage<H1, H2, 256, false>(sa1T, sw2, sc2, sh2, sa2T, nullptr, lane, wave);
    __syncthreads();

    // ---- stage 3: [128 x 64] -> pooled [64 x 128] (f32)
    conv_stage<H2, H3, 128, true>(sa2T, sw3, sc3, sh3, nullptr, sa3, lane, wave);
    __syncthreads();

    // ---- global average pool over 64 positions -> feat[n, 128] ---------
    for (int o = tid; o < H3; o += 256) {
        float s = 0.f;
        #pragma unroll
        for (int i = 0; i < 64; ++i) s += sa3[i * H3 + o];
        feat[(size_t)n * FDIM + o] = s * (1.0f / 64.0f);
    }
}

// ---------------------------------------------------------------------------
// Linear(128) -> BN -> ReLU -> L2 normalize. A = weight rows (F), B = samples.
// ---------------------------------------------------------------------------
__global__ __launch_bounds__(128)
void head_kernel(const float* __restrict__ feat,
                 const float* __restrict__ wf, const float* __restrict__ bfc,
                 const float* __restrict__ gf, const float* __restrict__ bef,
                 const float* __restrict__ mf, const float* __restrict__ vf,
                 float* __restrict__ featn)
{
    __shared__ __align__(16) __bf16 sA[16 * FDIM];       // samples  (B matrix)
    __shared__ __align__(16) __bf16 sW[FDIM * FDIM];     // weights  (A matrix)
    __shared__ __align__(16) float  sY[16 * FDIM];
    __shared__ float scf[FDIM], shf[FDIM], sSS[16];

    const int tid = threadIdx.x, lane = tid & 31, wave = tid >> 5;
    const int nlo = lane & 15, hi = lane >> 4;
    const int n0 = blockIdx.x * 16;

    for (int i = tid; i < 16 * FDIM;   i += 128) sA[i] = (__bf16)feat[(size_t)n0 * FDIM + i];
    for (int i = tid; i < FDIM * FDIM; i += 128) sW[i] = (__bf16)wf[i];
    for (int i = tid; i < FDIM; i += 128) {
        float sc = gf[i] * rsqrtf(vf[i] + EPS_BN);
        scf[i] = sc; shf[i] = (bfc[i] - mf[i]) * sc + bef[i];
    }
    if (tid < 16) sSS[tid] = 0.f;
    __syncthreads();

    for (int ft = wave; ft < 8; ft += 4) {
        const int frow = ft * 16 + nlo;                 // A row = output feature
        v8f acc = {0.f,0.f,0.f,0.f,0.f,0.f,0.f,0.f};
        #pragma unroll
        for (int ks = 0; ks < 4; ++ks) {
            const __bf16* wr = sW + frow * FDIM + ks * 32;
            const __bf16* br = sA + nlo  * FDIM + ks * 32;   // B col = sample
            bf16x8 a0 = *(const bf16x8*)(wr + hi * 8);
            bf16x8 a1 = *(const bf16x8*)(wr + 16 + hi * 8);
            bf16x8 b0 = *(const bf16x8*)(br + hi * 8);
            bf16x8 b1 = *(const bf16x8*)(br + 16 + hi * 8);
            acc = wmma_bf16(cat8(a0, a1), cat8(b0, b1), acc);
        }
        const int s  = nlo;                              // D: n = sample
        const int f0 = ft * 16 + 8 * hi;                 // D: m = feature (8 consec)
        float ss = 0.f, ys[8];
        #pragma unroll
        for (int j = 0; j < 8; ++j) {
            float y = fmaxf(acc[j] * scf[f0 + j] + shf[f0 + j], 0.0f);
            ys[j] = y; ss += y * y;
        }
        v4f lo = {ys[0], ys[1], ys[2], ys[3]};
        v4f hf = {ys[4], ys[5], ys[6], ys[7]};
        *(v4f*)(sY + s * FDIM + f0)     = lo;
        *(v4f*)(sY + s * FDIM + f0 + 4) = hf;
        atomicAdd(&sSS[s], ss);
    }
    __syncthreads();

    for (int i = tid; i < 16 * FDIM; i += 128) {
        int s = i >> 7;
        float nrm = fmaxf(sqrtf(sSS[s]), 1e-12f);
        featn[(size_t)n0 * FDIM + i] = sY[i] / nrm;
    }
}

// ---------------------------------------------------------------------------
// Class-mean prototypes per episode.
// ---------------------------------------------------------------------------
__global__ __launch_bounds__(128)
void proto_kernel(const float* __restrict__ featn, const int* __restrict__ slab,
                  float* __restrict__ protos)
{
    __shared__ float sp[NWAY * FDIM];
    __shared__ int   scnt[NWAY];
    const int tid = threadIdx.x;
    const int b   = blockIdx.x;

    for (int i = tid; i < NWAY * FDIM; i += 128) sp[i] = 0.f;
    if (tid < NWAY) {
        int c = 0;
        for (int s = 0; s < NSHOT; ++s) c += (slab[b * NSHOT + s] == tid);
        scnt[tid] = c;
    }
    __syncthreads();

    const int f = tid;                  // 128 threads == FDIM
    for (int s = 0; s < NSHOT; ++s) {
        int lab = slab[b * NSHOT + s];
        sp[lab * FDIM + f] += featn[((size_t)b * NSHOT + s) * FDIM + f];
    }
    __syncthreads();

    for (int w = 0; w < NWAY; ++w) {
        int c = scnt[w] > 0 ? scnt[w] : 1;
        protos[((size_t)b * NWAY + w) * FDIM + f] = sp[w * FDIM + f] / (float)c;
    }
}

// ---------------------------------------------------------------------------
// out[b,q,w] = -||qf - proto||_2
// ---------------------------------------------------------------------------
__global__ __launch_bounds__(128)
void dist_kernel(const float* __restrict__ featn, const float* __restrict__ protos,
                 float* __restrict__ out)
{
    const int tid = threadIdx.x, lane = tid & 31, wave = tid >> 5;
    const int bq = blockIdx.x;                 // b*NQRY + q
    const int b  = bq / NQRY;
    __shared__ float red[4];

    const float qv = featn[((size_t)NSUP + bq) * FDIM + tid];
    for (int w = 0; w < NWAY; ++w) {
        float d = qv - protos[((size_t)b * NWAY + w) * FDIM + tid];
        float s = d * d;
        #pragma unroll
        for (int off = 16; off > 0; off >>= 1) s += __shfl_down(s, off, 32);
        if (lane == 0) red[wave] = s;
        __syncthreads();
        if (tid == 0) {
            float d2 = red[0] + red[1] + red[2] + red[3];
            out[(size_t)bq * NWAY + w] = -sqrtf(fmaxf(d2, 0.0f));
        }
        __syncthreads();
    }
}

// ---------------------------------------------------------------------------
extern "C" void kernel_launch(void* const* d_in, const int* in_sizes, int n_in,
                              void* d_out, int out_size, void* d_ws, size_t ws_size,
                              hipStream_t stream)
{
    const float* s_img = (const float*)d_in[0];
    const float* q_img = (const float*)d_in[1];
    const int*   slab  = (const int*)  d_in[2];
    const float* w1 = (const float*)d_in[3];
    const float* b1 = (const float*)d_in[4];
    const float* g1 = (const float*)d_in[5];
    const float* be1= (const float*)d_in[6];
    const float* m1 = (const float*)d_in[7];
    const float* v1 = (const float*)d_in[8];
    const float* w2 = (const float*)d_in[9];
    const float* b2 = (const float*)d_in[10];
    const float* g2 = (const float*)d_in[11];
    const float* be2= (const float*)d_in[12];
    const float* m2 = (const float*)d_in[13];
    const float* v2 = (const float*)d_in[14];
    const float* w3 = (const float*)d_in[15];
    const float* b3 = (const float*)d_in[16];
    const float* g3 = (const float*)d_in[17];
    const float* be3= (const float*)d_in[18];
    const float* m3 = (const float*)d_in[19];
    const float* v3 = (const float*)d_in[20];
    const float* wf = (const float*)d_in[21];
    const float* bfc= (const float*)d_in[22];
    const float* gf = (const float*)d_in[23];
    const float* bef= (const float*)d_in[24];
    const float* mf = (const float*)d_in[25];
    const float* vf = (const float*)d_in[26];

    char* ws = (char*)d_ws;
    float* feat   = (float*)ws;                                      // 9600*128 f32
    float* featn  = (float*)(ws + (size_t)NSAMP * FDIM * 4);         // 9600*128 f32
    float* protos = (float*)(ws + (size_t)2 * NSAMP * FDIM * 4);     // 32*20*128 f32

    encoder_conv_kernel<<<NSAMP, 256, 0, stream>>>(
        s_img, q_img,
        w1, b1, g1, be1, m1, v1,
        w2, b2, g2, be2, m2, v2,
        w3, b3, g3, be3, m3, v3,
        feat);
    head_kernel<<<NSAMP / 16, 128, 0, stream>>>(feat, wf, bfc, gf, bef, mf, vf, featn);
    proto_kernel<<<B_EP, 128, 0, stream>>>(featn, slab, protos);
    dist_kernel<<<B_EP * NQRY, 128, 0, stream>>>(featn, protos, (float*)d_out);
}
